// ResGCNCell_70239895159467
// MI455X (gfx1250) — matrix operand order repaired and checked by reading, not compiled, not run on hardware
//
#include <hip/hip_runtime.h>
#include <stdint.h>

// ---------------- problem constants ----------------
#define BB   8
#define NNN  1024
#define DIN_ 512
#define DD   256

typedef __attribute__((ext_vector_type(16))) __bf16 bf16x16;
typedef __attribute__((ext_vector_type(8)))  __bf16 bf16x8;
typedef __attribute__((ext_vector_type(8)))  float  f32x8;

__device__ __forceinline__ uint16_t f2bf(float f) {
  uint32_t u = __float_as_uint(f);
  u += 0x7FFFu + ((u >> 16) & 1u);     // round-to-nearest-even
  return (uint16_t)(u >> 16);
}

__device__ __forceinline__ bf16x16 load_frag(const uint16_t* p0, const uint16_t* p1) {
  bf16x8 lo = *reinterpret_cast<const bf16x8*>(p0);
  bf16x8 hi = *reinterpret_cast<const bf16x8*>(p1);
  bf16x16 r;
#pragma unroll
  for (int i = 0; i < 8; ++i) { r[i] = lo[i]; r[i + 8] = hi[i]; }
  return r;
}

// ---------------- epilogue modes ----------------
enum { EPI_PROJ = 0, EPI_INPUTFC = 1, EPI_SCORE = 2, EPI_AGG = 3, EPI_FC = 4 };

// C = A(bf16)[M,K] @ Bsrc(bf16)[Ncols,K]^T (+bias), per-mode epilogue.
// Block: 256 threads = 8 waves; block tile 128x128; wave tile 32x64
// (2 A-frags x 4 B-frags -> 8 WMMA per K-step, 1.5 ds_b128/WMMA).
// Double-buffered LDS, register-prefetched global tiles, two barriers per step.
#define LDSK 40   // LDS row stride (elements); 80B = multiple of 16B
template <int EPI>
__global__ __launch_bounds__(256) void gemm_wmma(
    const uint16_t* __restrict__ A, const uint16_t* __restrict__ Bsrc,
    float* __restrict__ Cf, uint16_t* __restrict__ Cb,
    const float* __restrict__ bias,
    const uint32_t* __restrict__ maskJ,   // EPI_SCORE only: [B*N][32] bit-mask
    int K, int lda, int ldb, int ldc,
    long sA, long sB, long sC) {
  constexpr bool kF32  = (EPI == EPI_INPUTFC || EPI == EPI_SCORE || EPI == EPI_AGG || EPI == EPI_FC);
  constexpr bool kBF16 = (EPI == EPI_PROJ || EPI == EPI_INPUTFC || EPI == EPI_AGG);
  constexpr bool kBias = (EPI == EPI_PROJ || EPI == EPI_INPUTFC || EPI == EPI_FC);

  const int bz = blockIdx.z;
  A    += (long)bz * sA;
  Bsrc += (long)bz * sB;
  const int mBase = blockIdx.y * 128;
  const int nBase = blockIdx.x * 128;

  __shared__ uint16_t ldsA[2][128 * LDSK];
  __shared__ uint16_t ldsB[2][128 * LDSK];

  const int tid  = threadIdx.x;
  const int w    = tid >> 5;
  const int lane = tid & 31;
  const int lrow = lane & 15;
  const int kh   = lane >> 4;
  const int wr   = w >> 1;      // wave row (0..3) -> rows 32*wr
  const int wc   = w & 1;       // wave col (0..1) -> cols 64*wc

  const int sr   = tid >> 1;    // staging row 0..127
  const int shalf= tid & 1;     // staging 16-elem half

  f32x8 acc[2][4];
#pragma unroll
  for (int a = 0; a < 2; ++a)
#pragma unroll
    for (int b = 0; b < 4; ++b)
#pragma unroll
      for (int i = 0; i < 8; ++i) acc[a][b][i] = 0.0f;

  uint4 aq0, aq1, bq0, bq1;
  {
    const uint4* sa = reinterpret_cast<const uint4*>(A + (long)(mBase + sr) * lda + shalf * 16);
    aq0 = sa[0]; aq1 = sa[1];
    const uint4* sb = reinterpret_cast<const uint4*>(Bsrc + (long)(nBase + sr) * ldb + shalf * 16);
    bq0 = sb[0]; bq1 = sb[1];
  }

  int buf = 0;
  for (int k0 = 0; k0 < K; k0 += 32) {
    { // commit staged regs to LDS
      uint4* da = reinterpret_cast<uint4*>(&ldsA[buf][sr * LDSK + shalf * 16]);
      da[0] = aq0; da[1] = aq1;
      uint4* db = reinterpret_cast<uint4*>(&ldsB[buf][sr * LDSK + shalf * 16]);
      db[0] = bq0; db[1] = bq1;
    }
    __syncthreads();
    if (k0 + 32 < K) { // prefetch next K tile into regs
      const uint4* sa = reinterpret_cast<const uint4*>(A + (long)(mBase + sr) * lda + (k0 + 32) + shalf * 16);
      aq0 = sa[0]; aq1 = sa[1];
      const uint4* sb = reinterpret_cast<const uint4*>(Bsrc + (long)(nBase + sr) * ldb + (k0 + 32) + shalf * 16);
      bq0 = sb[0]; bq1 = sb[1];
    }
    // A frag rows: 32*wr + 16*aa + lrow ; element e -> K = (e<8?e:e+8)+8*kh
    bf16x16 af[2], bfr[4];
#pragma unroll
    for (int aa = 0; aa < 2; ++aa) {
      const uint16_t* p = &ldsA[buf][(wr * 32 + aa * 16 + lrow) * LDSK + kh * 8];
      af[aa] = load_frag(p, p + 16);
    }
    // B frag cols: 64*wc + 16*bb + lrow ; element e -> K = (e<8?e:e+8)+8*kh
#pragma unroll
    for (int bb = 0; bb < 4; ++bb) {
      const uint16_t* p = &ldsB[buf][(wc * 64 + bb * 16 + lrow) * LDSK + kh * 8];
      bfr[bb] = load_frag(p, p + 16);
    }
#pragma unroll
    for (int aa = 0; aa < 2; ++aa)
#pragma unroll
      for (int bb = 0; bb < 4; ++bb)
        acc[aa][bb] = __builtin_amdgcn_wmma_f32_16x16x32_bf16(
            false, af[aa], false, bfr[bb], (short)0, acc[aa][bb], false, false);
    buf ^= 1;
    __syncthreads();
  }

  // epilogue: m = mBase + 32*wr + 16*aa + 8*kh + i ; n = nBase + 64*wc + 16*bb + lrow
#pragma unroll
  for (int aa = 0; aa < 2; ++aa) {
#pragma unroll
    for (int bb = 0; bb < 4; ++bb) {
      int n = nBase + wc * 64 + bb * 16 + lrow;
      float bv = kBias ? bias[n] : 0.0f;
#pragma unroll
      for (int i = 0; i < 8; ++i) {
        int m = mBase + wr * 32 + aa * 16 + kh * 8 + i;
        float v = acc[aa][bb][i] + bv;
        if constexpr (EPI == EPI_SCORE) {
          v = (v > 0.0f) ? v : 0.01f * v;  // leaky_relu(0.01)
          uint32_t word = maskJ[((long)bz * NNN + m) * 32 + (n >> 5)];
          v = ((word >> (n & 31)) & 1u) ? v : -1e9f;
        }
        long ci = (long)bz * sC + (long)m * ldc + n;
        if constexpr (kF32)  Cf[ci] = v;
        if constexpr (kBF16) Cb[ci] = f2bf(v);
      }
    }
  }
}

// ---------------- setup / elementwise kernels ----------------
__global__ __launch_bounds__(256) void f2bf_kernel(const float* __restrict__ in,
                                                   uint16_t* __restrict__ out, long n) {
  long i = (long)blockIdx.x * 256 + threadIdx.x;
  long stride = (long)gridDim.x * 256;
  for (; i < n; i += stride) out[i] = f2bf(in[i]);
}

// Batched f32 [b][R][C] -> bf16 [b][C][R] transpose via 32x32 LDS tile.
__global__ __launch_bounds__(256) void transpose_f2bf(const float* __restrict__ in,
                                                      uint16_t* __restrict__ out,
                                                      int R, int C) {
  __shared__ float t[32][33];
  const int tx = threadIdx.x & 31, ty = threadIdx.x >> 5;  // 32 x 8
  const int r0 = blockIdx.y * 32, c0 = blockIdx.x * 32;
  const long base = (long)blockIdx.z * R * C;
#pragma unroll
  for (int p = 0; p < 4; ++p)
    t[ty + p * 8][tx] = in[base + (long)(r0 + ty + p * 8) * C + c0 + tx];
  __syncthreads();
#pragma unroll
  for (int p = 0; p < 4; ++p)
    out[base + (long)(c0 + ty + p * 8) * R + r0 + tx] = f2bf(t[tx][ty + p * 8]);
}

// mask[dir][b*N+q][w] bit n: (adj[b][q][n]==dir+1) || (row empty && n==q)
__global__ __launch_bounds__(128) void maskpack_kernel(const int* __restrict__ adj,
                                                       uint32_t* __restrict__ mb) {
  const long row = blockIdx.x;          // b*N + q
  const int q = (int)(row & (NNN - 1));
  const int* a = adj + row * NNN;
  __shared__ uint32_t w1[32], w2[32];
  const int t = threadIdx.x;
  if (t < 32) { w1[t] = 0u; w2[t] = 0u; }
  __syncthreads();
  uint32_t m1 = 0, m2 = 0;
#pragma unroll
  for (int i = 0; i < 8; ++i) {
    int v = a[t * 8 + i];
    m1 |= (v == 1) ? (1u << i) : 0u;
    m2 |= (v == 2) ? (1u << i) : 0u;
  }
  atomicOr(&w1[t >> 2], m1 << ((t & 3) * 8));
  atomicOr(&w2[t >> 2], m2 << ((t & 3) * 8));
  __syncthreads();
  if (t == 0) {
    uint32_t any1 = 0, any2 = 0;
    for (int i = 0; i < 32; ++i) { any1 |= w1[i]; any2 |= w2[i]; }
    if (any1 == 0) w1[q >> 5] |= 1u << (q & 31);   // self loop
    if (any2 == 0) w2[q >> 5] |= 1u << (q & 31);
  }
  __syncthreads();
  if (t < 32) {
    mb[row * 32 + t]                          = w1[t];
    mb[((long)BB * NNN + row) * 32 + t]       = w2[t];
  }
}

// row softmax over 1024 cols, f32 in -> bf16 out
__global__ __launch_bounds__(256) void softmax_rows(const float* __restrict__ S,
                                                    uint16_t* __restrict__ P) {
  long row = blockIdx.x;
  const float* s = S + row * NNN;
  uint16_t* p = P + row * NNN;
  int t = threadIdx.x;
  __shared__ float red[256];
  float v[4], mx = -3.0e38f;
#pragma unroll
  for (int i = 0; i < 4; ++i) { v[i] = s[t + i * 256]; mx = fmaxf(mx, v[i]); }
  red[t] = mx; __syncthreads();
  for (int st = 128; st > 0; st >>= 1) { if (t < st) red[t] = fmaxf(red[t], red[t + st]); __syncthreads(); }
  mx = red[0]; __syncthreads();
  float sum = 0.0f;
#pragma unroll
  for (int i = 0; i < 4; ++i) { v[i] = __expf(v[i] - mx); sum += v[i]; }
  red[t] = sum; __syncthreads();
  for (int st = 128; st > 0; st >>= 1) { if (t < st) red[t] += red[t + st]; __syncthreads(); }
  float inv = 1.0f / red[0];
#pragma unroll
  for (int i = 0; i < 4; ++i) p[t + i * 256] = f2bf(v[i] * inv);
}

__device__ __forceinline__ float ln256(float v, float g, float b, float* red, int d) {
  red[d] = v; __syncthreads();
  for (int st = 128; st > 0; st >>= 1) { if (d < st) red[d] += red[d + st]; __syncthreads(); }
  float mean = red[0] * (1.0f / 256.0f); __syncthreads();
  red[d] = v * v; __syncthreads();
  for (int st = 128; st > 0; st >>= 1) { if (d < st) red[d] += red[d + st]; __syncthreads(); }
  float var = red[0] * (1.0f / 256.0f) - mean * mean;
  return (v - mean) * rsqrtf(var + 1e-5f) * g + b;
}

// out = LayerNorm(relu(softmax(aggre) . [o0,o1])) ; dual-store
__global__ __launch_bounds__(256) void agg_ln_kernel(
    const float* __restrict__ o0, const float* __restrict__ o1,
    const float* __restrict__ ag, const float* __restrict__ g, const float* __restrict__ b,
    float* __restrict__ of, uint16_t* __restrict__ ob) {
  long row = blockIdx.x; int d = threadIdx.x;
  float a0 = ag[0], a1 = ag[1];
  float m = fmaxf(a0, a1);
  float e0 = __expf(a0 - m), e1 = __expf(a1 - m);
  float w0 = e0 / (e0 + e1), w1 = e1 / (e0 + e1);
  float v = fmaxf(w0 * o0[row * DD + d] + w1 * o1[row * DD + d], 0.0f);
  __shared__ float red[256];
  float y = ln256(v, g[d], b[d], red, d);
  of[row * DD + d] = y;
  ob[row * DD + d] = f2bf(y);
}

// out = LayerNorm(relu(in)) ; dual-store
__global__ __launch_bounds__(256) void relu_ln_kernel(
    const float* __restrict__ in, const float* __restrict__ g, const float* __restrict__ b,
    float* __restrict__ of, uint16_t* __restrict__ ob) {
  long row = blockIdx.x; int d = threadIdx.x;
  float v = fmaxf(in[row * DD + d], 0.0f);
  __shared__ float red[256];
  float y = ln256(v, g[d], b[d], red, d);
  of[row * DD + d] = y;
  ob[row * DD + d] = f2bf(y);
}

__global__ __launch_bounds__(256) void resid_kernel(float* __restrict__ x,
                                                    const float* __restrict__ inp,
                                                    uint16_t* __restrict__ xb) {
  long i = (long)blockIdx.x * 256 + threadIdx.x;
  float v = x[i] + inp[i];
  x[i] = v;
  xb[i] = f2bf(v);
}

// ---------------- host orchestration ----------------
extern "C" void kernel_launch(void* const* d_in, const int* in_sizes, int n_in,
                              void* d_out, int out_size, void* d_ws, size_t ws_size,
                              hipStream_t stream) {
  (void)in_sizes; (void)n_in; (void)out_size; (void)ws_size;
  const int*   adj   = (const int*)  d_in[0];
  const float* h     = (const float*)d_in[1];
  const float* ifw   = (const float*)d_in[2];
  const float* ifb   = (const float*)d_in[3];
  const float* aw    = (const float*)d_in[4];
  const float* ab    = (const float*)d_in[5];
  const float* aggre = (const float*)d_in[6];
  const float* cng   = (const float*)d_in[7];
  const float* cnb   = (const float*)d_in[8];
  const float* fw    = (const float*)d_in[9];
  const float* fb    = (const float*)d_in[10];
  const float* ng    = (const float*)d_in[11];
  const float* nb    = (const float*)d_in[12];

  char* ws = (char*)d_ws;
  size_t off = 0;
  auto alloc = [&](size_t bytes) -> char* {
    char* p = ws + off;
    off += (bytes + 255) & ~(size_t)255;
    return p;
  };
  const long MN = (long)BB * NNN;        // 8192 rows
  float*    x_f    = (float*)   alloc(MN * DD * 4);
  uint16_t* x_bf   = (uint16_t*)alloc(MN * DD * 2);
  uint16_t* xT_bf  = (uint16_t*)alloc(MN * DD * 2);   // [b][d][node]
  float*    inp_f  = (float*)   alloc(MN * DD * 4);
  float*    o0_f   = (float*)   alloc(MN * DD * 4);
  uint16_t* o0_bf  = (uint16_t*)alloc(MN * DD * 2);
  uint16_t* o0T_bf = (uint16_t*)alloc(MN * DD * 2);
  float*    o1_f   = (float*)   alloc(MN * DD * 4);
  uint16_t* o1_bf  = (uint16_t*)alloc(MN * DD * 2);
  uint16_t* proj_bf= (uint16_t*)alloc(MN * DD * 2);
  float*    S_f    = (float*)   alloc(MN * NNN * 4);
  uint16_t* P_bf   = (uint16_t*)alloc(MN * NNN * 2);
  float*    agg_f  = (float*)   alloc(MN * DD * 4);
  uint16_t* agg_bf = (uint16_t*)alloc(MN * DD * 2);
  uint16_t* h_bf   = (uint16_t*)alloc(MN * DIN_ * 2);
  uint16_t* ifwT   = (uint16_t*)alloc((long)DIN_ * DD * 2);      // [256][512]
  uint16_t* awT    = (uint16_t*)alloc((long)8 * DD * DD * 2);    // [wi][256][256]^T
  uint16_t* fwT    = (uint16_t*)alloc((long)8 * DD * DD * 2);
  uint32_t* mb     = (uint32_t*)alloc((long)2 * MN * 32 * 4);    // packed masks
  float*    y_f    = S_f;  // alias: FC pre-activation reuses score buffer

  // ---- setup: conversions, transposed weights, packed masks ----
  f2bf_kernel<<<4096, 256, 0, stream>>>(h, h_bf, MN * DIN_);
  { dim3 g(DD / 32, DIN_ / 32, 1); transpose_f2bf<<<g, 256, 0, stream>>>(ifw, ifwT, DIN_, DD); }
  { dim3 g(DD / 32, DD / 32, 8);   transpose_f2bf<<<g, 256, 0, stream>>>(aw,  awT,  DD,   DD); }
  { dim3 g(DD / 32, DD / 32, 8);   transpose_f2bf<<<g, 256, 0, stream>>>(fw,  fwT,  DD,   DD); }
  maskpack_kernel<<<MN, 128, 0, stream>>>(adj, mb);

  // x = h @ input_fc_w + b   (M=8192, K=512, N=256); B^T = ifwT [256][512]
  {
    dim3 g(DD / 128, MN / 128, 1);
    gemm_wmma<EPI_INPUTFC><<<g, 256, 0, stream>>>(
        h_bf, ifwT, x_f, x_bf, ifb, nullptr, DIN_, DIN_, DIN_, DD, 0, 0, 0);
  }

  for (int blk = 0; blk < 4; ++blk) {
    hipMemcpyAsync(inp_f, x_f, MN * DD * sizeof(float), hipMemcpyDeviceToDevice, stream);
    for (int hd = 0; hd < 2; ++hd) {
      const int wi = blk * 2 + hd;
      // transposed copy of x for this head's first aggregation
      { dim3 g(DD / 32, NNN / 32, BB); transpose_f2bf<<<g, 256, 0, stream>>>(x_f, xT_bf, NNN, DD); }
      const uint16_t* cur_bf = x_bf;
      const uint16_t* curT   = xT_bf;
      for (int j = 0; j < 2; ++j) {
        // proj = cur @ attn_w + attn_b  (per batch 1024x256, K=256); B^T = awT
        dim3 gp(DD / 128, NNN / 128, BB);
        gemm_wmma<EPI_PROJ><<<gp, 256, 0, stream>>>(
            cur_bf, awT + (long)wi * DD * DD, nullptr, proj_bf, ab + wi * DD,
            nullptr, DD, DD, DD, DD, (long)NNN * DD, 0, (long)NNN * DD);
        // S = mask_j(leaky_relu(proj @ cur^T))  (1024x1024, K=256)
        dim3 gs(NNN / 128, NNN / 128, BB);
        gemm_wmma<EPI_SCORE><<<gs, 256, 0, stream>>>(
            proj_bf, cur_bf, S_f, nullptr, nullptr, mb + (long)j * MN * 32,
            DD, DD, DD, NNN, (long)NNN * DD, (long)NNN * DD, (long)NNN * NNN);
        // P = softmax(S) rows
        softmax_rows<<<MN, 256, 0, stream>>>(S_f, P_bf);
        // cur' = P @ cur  (1024x256, K=1024); B^T = curT [d][node]
        float*    oF = (j == 0) ? o0_f  : o1_f;
        uint16_t* oB = (j == 0) ? o0_bf : o1_bf;
        dim3 ga(DD / 128, NNN / 128, BB);
        gemm_wmma<EPI_AGG><<<ga, 256, 0, stream>>>(
            P_bf, curT, oF, oB, nullptr, nullptr,
            NNN, NNN, NNN, DD, (long)NNN * NNN, (long)NNN * DD, (long)NNN * DD);
        if (j == 0) {
          dim3 gt(DD / 32, NNN / 32, BB);
          transpose_f2bf<<<gt, 256, 0, stream>>>(o0_f, o0T_bf, NNN, DD);
          cur_bf = o0_bf; curT = o0T_bf;
        }
      }
      // head epilogue: weighted aggregate + relu + LN(cn)
      agg_ln_kernel<<<MN, 256, 0, stream>>>(
          o0_f, o1_f, aggre + wi * 2, cng + wi * DD, cnb + wi * DD, agg_f, agg_bf);
      // fc: y = agg @ fc_w + fc_b (M=8192, K=N=256); B^T = fwT
      dim3 gf(DD / 128, MN / 128, 1);
      gemm_wmma<EPI_FC><<<gf, 256, 0, stream>>>(
          agg_bf, fwT + (long)wi * DD * DD, y_f, nullptr, fb + wi * DD,
          nullptr, DD, DD, DD, DD, 0, 0, 0);
      // x = LN(relu(y), n)
      relu_ln_kernel<<<MN, 256, 0, stream>>>(
          y_f, ng + wi * DD, nb + wi * DD, x_f, x_bf);
    }
    // block residual
    resid_kernel<<<(MN * DD) / 256, 256, 0, stream>>>(x_f, inp_f, x_bf);
  }

  hipMemcpyAsync(d_out, x_f, MN * DD * sizeof(float), hipMemcpyDeviceToDevice, stream);
}
